// QuantizedLinear_11123965297141
// MI455X (gfx1250) — matrix operand order repaired and checked by reading, not compiled, not run on hardware
//
#include <hip/hip_runtime.h>

typedef unsigned short ushort_t;
typedef unsigned int uint_t;

typedef __attribute__((ext_vector_type(16))) __bf16 v16bf;
typedef __attribute__((ext_vector_type(8)))  float  v8f;

static constexpr int TOKENS = 4096;
static constexpr int IN     = 4096;
static constexpr int OUT    = 4096;
static constexpr int RANK   = 64;

// ---------- bf16 helpers (RNE) ----------
__device__ __forceinline__ ushort_t f32_to_bf16(float f) {
    uint_t u = __float_as_uint(f);
    uint_t lsb = (u >> 16) & 1u;
    u += 0x7fffu + lsb;
    return (ushort_t)(u >> 16);
}
__device__ __forceinline__ float bf16_to_f32(ushort_t h) {
    return __uint_as_float(((uint_t)h) << 16);
}

// ---------- CDNA5 async global->LDS copy (ASYNCcnt path), with safe fallback ----------
// Per hipcc diagnostic, the builtin takes (AS1 int-vec16B*, AS3 ..., imm, imm).
typedef int v4i_t __attribute__((vector_size(16)));
typedef __attribute__((address_space(1))) v4i_t gv4i_t;
typedef __attribute__((address_space(3))) v4i_t lv4i_t;

__device__ __forceinline__ void async_copy_b128(const ushort_t* g, ushort_t* l) {
#if __has_builtin(__builtin_amdgcn_global_load_async_to_lds_b128)
    __builtin_amdgcn_global_load_async_to_lds_b128((gv4i_t*)g, (lv4i_t*)l, 0, 0);
#else
    *(uint4*)l = *(const uint4*)g;
#endif
}
__device__ __forceinline__ void wait_async_copies() {
#if __has_builtin(__builtin_amdgcn_s_wait_asynccnt)
    __builtin_amdgcn_s_wait_asynccnt(0);
#elif __has_builtin(__builtin_amdgcn_global_load_async_to_lds_b128)
    asm volatile("s_wait_asynccnt 0" ::: "memory");
#endif
}

// ---------- kernel 1: split x into bf16 hi + bf16 lo ----------
__global__ __launch_bounds__(256) void split_x_kernel(
    const float* __restrict__ x, ushort_t* __restrict__ xh, ushort_t* __restrict__ xl)
{
    int i = blockIdx.x * 256 + threadIdx.x;           // index into float4 view
    const int n4 = (TOKENS * IN) / 4;
    if (i >= n4) return;
    float4 v = ((const float4*)x)[i];
    ushort_t h0 = f32_to_bf16(v.x), h1 = f32_to_bf16(v.y);
    ushort_t h2 = f32_to_bf16(v.z), h3 = f32_to_bf16(v.w);
    ushort_t l0 = f32_to_bf16(v.x - bf16_to_f32(h0));
    ushort_t l1 = f32_to_bf16(v.y - bf16_to_f32(h1));
    ushort_t l2 = f32_to_bf16(v.z - bf16_to_f32(h2));
    ushort_t l3 = f32_to_bf16(v.w - bf16_to_f32(h3));
    uint2 hv, lv;
    hv.x = (uint_t)h0 | ((uint_t)h1 << 16);
    hv.y = (uint_t)h2 | ((uint_t)h3 << 16);
    lv.x = (uint_t)l0 | ((uint_t)l1 << 16);
    lv.y = (uint_t)l2 | ((uint_t)l3 << 16);
    ((uint2*)xh)[i] = hv;
    ((uint2*)xl)[i] = lv;
}

// ---------- kernel 2: dequantize weight (gather + affine + rank-64) -> bf16 hi/lo ----------
// w[o][i] = centroids[indices[o][perm[i]>>3]][perm[i]&7]*scale[o] + wbias[o]
//           + sum_r (U[o][r]*S[r]) * Vt[r][i]
__global__ __launch_bounds__(256) void dequant_kernel(
    const float* __restrict__ centroids, const int* __restrict__ indices,
    const int* __restrict__ perm, const float* __restrict__ wscale,
    const float* __restrict__ wbias, const float* __restrict__ U,
    const float* __restrict__ S, const float* __restrict__ Vt,
    ushort_t* __restrict__ wh, ushort_t* __restrict__ wl)
{
    __shared__ float us[64][RANK + 1];   // (o_local, r)   U*S
    __shared__ float vt[RANK][64 + 1];   // (r, i_local)

    const int o0 = blockIdx.y * 64;
    const int i0 = blockIdx.x * 64;
    const int tid = threadIdx.x;

    for (int j = tid; j < 64 * RANK; j += 256) {
        int ol = j >> 6, r = j & 63;
        us[ol][r] = U[(size_t)(o0 + ol) * RANK + r] * S[r];
    }
    for (int j = tid; j < RANK * 64; j += 256) {
        int r = j >> 6, il = j & 63;
        vt[r][il] = Vt[(size_t)r * IN + i0 + il];
    }
    __syncthreads();

    const int ol = tid >> 2;            // 0..63
    const int ig = (tid & 3) << 4;      // 0,16,32,48
    const int o  = o0 + ol;
    const float sc = wscale[o];
    const float wb = wbias[o];
    const int* indrow = indices + (size_t)o * (IN / 8);

    float acc[16];
    #pragma unroll
    for (int j = 0; j < 16; ++j) {
        int i = i0 + ig + j;
        int p = perm[i];
        int idx = indrow[p >> 3];
        float c = centroids[(size_t)idx * 8 + (p & 7)];
        acc[j] = c * sc + wb;
    }
    #pragma unroll 4
    for (int r = 0; r < RANK; ++r) {
        float uo = us[ol][r];
        #pragma unroll
        for (int j = 0; j < 16; ++j) acc[j] += uo * vt[r][ig + j];
    }

    union Pack { uint4 q[2]; ushort_t u[16]; } ph, pl;
    #pragma unroll
    for (int j = 0; j < 16; ++j) {
        ushort_t h = f32_to_bf16(acc[j]);
        ph.u[j] = h;
        pl.u[j] = f32_to_bf16(acc[j] - bf16_to_f32(h));
    }
    size_t base = (size_t)o * IN + i0 + ig;
    ((uint4*)(wh + base))[0] = ph.q[0];
    ((uint4*)(wh + base))[1] = ph.q[1];
    ((uint4*)(wl + base))[0] = pl.q[0];
    ((uint4*)(wl + base))[1] = pl.q[1];
}

// ---------- kernel 3: LDS-staged GEMM, v_wmma_f32_16x16x32_bf16, hi/lo 3-product ----------
// out[t][o] = sum_i x[t][i]*w[o][i] + bias[o]
// Block: 256 thr = 8 waves; block tile 128(M) x 128(N), K-step 32.
// Wave grid 2(M) x 4(N); wave tile 64 x 32 = 4x2 WMMA accumulators.
// Tiles double-buffered in LDS, filled via async global->LDS copies (ASYNCcnt).
union FragU { v16bf v; uint4 q[2]; };

static constexpr int LDSTRIDE = 40;   // 32 K-elems + 8 pad (bf16); 80 B row stride, 16B aligned

__global__ __launch_bounds__(256) void gemm_kernel(
    const ushort_t* __restrict__ xh, const ushort_t* __restrict__ xl,
    const ushort_t* __restrict__ wh, const ushort_t* __restrict__ wl,
    const float* __restrict__ bias, float* __restrict__ out)
{
    // lds[buf][arr][row*LDSTRIDE + k]; arr: 0=Ah 1=Al 2=Bh 3=Bl
    __shared__ ushort_t lds[2][4][128 * LDSTRIDE];

    const int tid  = threadIdx.x;
    const int lane = tid & 31;
    const int wave = tid >> 5;                 // 0..7
    const int wm = wave >> 2;                  // 0..1  -> M offset wm*64
    const int wn = wave & 3;                   // 0..3  -> N offset wn*32
    const int mBase = blockIdx.y * 128;        // token rows
    const int nBase = blockIdx.x * 128;        // output cols

    const int half = lane >> 4;                // 0/1
    const int l15  = lane & 15;

    // tile-copy assignment: thread t handles row t>>1, 32B chunk (t&1)
    const int crow = tid >> 1;
    const int ckof = (tid & 1) * 16;
    const ushort_t* gA_h = xh + (size_t)(mBase + crow) * IN + ckof;
    const ushort_t* gA_l = xl + (size_t)(mBase + crow) * IN + ckof;
    const ushort_t* gB_h = wh + (size_t)(nBase + crow) * IN + ckof;
    const ushort_t* gB_l = wl + (size_t)(nBase + crow) * IN + ckof;
    const int ldsOff = crow * LDSTRIDE + ckof;

    auto load_tile = [&](int buf, int k) {
        async_copy_b128(gA_h + k,     &lds[buf][0][ldsOff]);
        async_copy_b128(gA_h + k + 8, &lds[buf][0][ldsOff + 8]);
        async_copy_b128(gA_l + k,     &lds[buf][1][ldsOff]);
        async_copy_b128(gA_l + k + 8, &lds[buf][1][ldsOff + 8]);
        async_copy_b128(gB_h + k,     &lds[buf][2][ldsOff]);
        async_copy_b128(gB_h + k + 8, &lds[buf][2][ldsOff + 8]);
        async_copy_b128(gB_l + k,     &lds[buf][3][ldsOff]);
        async_copy_b128(gB_l + k + 8, &lds[buf][3][ldsOff + 8]);
    };

    v8f acc[4][2];
    #pragma unroll
    for (int mi = 0; mi < 4; ++mi)
        #pragma unroll
        for (int ni = 0; ni < 2; ++ni)
            acc[mi][ni] = (v8f)(0.0f);

    // fragment source rows inside LDS (contiguous K order, same for A and B):
    // A: lane holds row m = wm*64 + mi*16 + l15; K = half*16 + {0..15}
    // B: lane holds col o = wn*32 + ni*16 + l15; K = half*16 + {0..15}
    const int aRow = wm * 64 + l15;
    const int bRow = wn * 32 + l15;
    const int aOff = aRow * LDSTRIDE + half * 16;
    const int bOff = bRow * LDSTRIDE + half * 16;

    load_tile(0, 0);
    wait_async_copies();
    __syncthreads();

    int buf = 0;
    for (int k = 0; k < IN; k += 32) {
        const int nbuf = buf ^ 1;
        if (k + 32 < IN) load_tile(nbuf, k + 32);

        const ushort_t* Ah = lds[buf][0];
        const ushort_t* Al = lds[buf][1];
        const ushort_t* Bh = lds[buf][2];
        const ushort_t* Bl = lds[buf][3];

        FragU ahf[4], alf[4], bhf[2], blf[2];
        #pragma unroll
        for (int mi = 0; mi < 4; ++mi) {
            const ushort_t* p = Ah + aOff + mi * 16 * LDSTRIDE;
            ahf[mi].q[0] = *(const uint4*)(p);
            ahf[mi].q[1] = *(const uint4*)(p + 8);
            const ushort_t* q = Al + aOff + mi * 16 * LDSTRIDE;
            alf[mi].q[0] = *(const uint4*)(q);
            alf[mi].q[1] = *(const uint4*)(q + 8);
        }
        #pragma unroll
        for (int ni = 0; ni < 2; ++ni) {
            const ushort_t* p = Bh + bOff + ni * 16 * LDSTRIDE;
            bhf[ni].q[0] = *(const uint4*)(p);
            bhf[ni].q[1] = *(const uint4*)(p + 8);
            const ushort_t* q = Bl + bOff + ni * 16 * LDSTRIDE;
            blf[ni].q[0] = *(const uint4*)(q);
            blf[ni].q[1] = *(const uint4*)(q + 8);
        }

        #pragma unroll
        for (int mi = 0; mi < 4; ++mi) {
            #pragma unroll
            for (int ni = 0; ni < 2; ++ni) {
                acc[mi][ni] = __builtin_amdgcn_wmma_f32_16x16x32_bf16(
                    false, ahf[mi].v, false, bhf[ni].v, (short)0, acc[mi][ni], false, false);
                acc[mi][ni] = __builtin_amdgcn_wmma_f32_16x16x32_bf16(
                    false, ahf[mi].v, false, blf[ni].v, (short)0, acc[mi][ni], false, false);
                acc[mi][ni] = __builtin_amdgcn_wmma_f32_16x16x32_bf16(
                    false, alf[mi].v, false, bhf[ni].v, (short)0, acc[mi][ni], false, false);
            }
        }

        wait_async_copies();   // our async writes into nbuf have landed
        __syncthreads();       // everyone done reading buf & writing nbuf
        buf = nbuf;
    }

    // C/D layout: VGPR v -> M = v + half*8, N = l15
    #pragma unroll
    for (int mi = 0; mi < 4; ++mi) {
        #pragma unroll
        for (int ni = 0; ni < 2; ++ni) {
            const int col = nBase + wn * 32 + ni * 16 + l15;
            const float b = bias[col];
            #pragma unroll
            for (int v = 0; v < 8; ++v) {
                const int row = mBase + wm * 64 + mi * 16 + half * 8 + v;
                out[(size_t)row * OUT + col] = acc[mi][ni][v] + b;
            }
        }
    }
}

// ---------- launcher ----------
extern "C" void kernel_launch(void* const* d_in, const int* in_sizes, int n_in,
                              void* d_out, int out_size, void* d_ws, size_t ws_size,
                              hipStream_t stream) {
    const float* x         = (const float*)d_in[0];
    const float* centroids = (const float*)d_in[1];
    const int*   indices   = (const int*)d_in[2];
    const int*   perm      = (const int*)d_in[3];
    const float* wscale    = (const float*)d_in[4];
    const float* wbias     = (const float*)d_in[5];
    const float* U         = (const float*)d_in[6];
    const float* S         = (const float*)d_in[7];
    const float* Vt        = (const float*)d_in[8];
    const float* bias      = (const float*)d_in[9];
    float* out = (float*)d_out;

    // workspace layout: x_hi, x_lo, w_hi, w_lo  (bf16 each, 32 MB each = 128 MB)
    ushort_t* xh = (ushort_t*)d_ws;
    ushort_t* xl = xh + (size_t)TOKENS * IN;
    ushort_t* wh = xl + (size_t)TOKENS * IN;
    ushort_t* wl = wh + (size_t)OUT * IN;

    split_x_kernel<<<(TOKENS * IN / 4 + 255) / 256, 256, 0, stream>>>(x, xh, xl);

    dequant_kernel<<<dim3(IN / 64, OUT / 64), 256, 0, stream>>>(
        centroids, indices, perm, wscale, wbias, U, S, Vt, wh, wl);

    gemm_kernel<<<dim3(OUT / 128, TOKENS / 128), 256, 0, stream>>>(
        xh, xl, wh, wl, bias, out);
}
